// Deform_block_86165633893045
// MI455X (gfx1250) — matrix operand order repaired and checked by reading, not compiled, not run on hardware
//
#include <hip/hip_runtime.h>
#include <hip/hip_bf16.h>
#include <math.h>

typedef __attribute__((ext_vector_type(16))) _Float16 v16h;
typedef __attribute__((ext_vector_type(8)))  _Float16 v8h;
typedef __attribute__((ext_vector_type(8)))  float    v8f;

#define BATCH 4
#define CIN   64
#define COUT  64
#define NFEAT 64
#define HH    128
#define WW    128
#define KTAP  9
#define NPIX  (BATCH * HH * WW)     // 65536
#define KC    (CIN * KTAP)          // 576  (both GEMMs have K=576 since NF==C==64)
#define MCONV (CIN * 3 * KTAP)      // 1728 offset/mask channels
#define OFFCH (CIN * 2 * KTAP)      // 1152 offset channels (rest are masks)

// ---------------------------------------------------------------------------
// WMMA fragment loaders (wave32, 16x16x32 f16; contiguous k order used by this code for both A and B)
// A (16x32): lane m=lane&15; halves e = K kbase+e, kbase=(lane>=16?16:0)
//            -> 16 contiguous halves from [M][K] rows = two b128 loads
// B (32x16): lane n=lane&15; halves e = K kbase+e, kbase=(lane>=16?16:0)
//            -> 16 contiguous halves from [N][K] rows = two b128 loads
// ---------------------------------------------------------------------------
__device__ __forceinline__ v16h frag_a(const _Float16* __restrict__ p) {
  v8h lo = *(const v8h*)(p);
  v8h hi = *(const v8h*)(p + 8);
  v16h r;
#pragma unroll
  for (int i = 0; i < 8; ++i) { r[i] = lo[i]; r[i + 8] = hi[i]; }
  return r;
}

__device__ __forceinline__ v16h frag_b(const _Float16* __restrict__ p) {
  v8h lo = *(const v8h*)(p);
  v8h hi = *(const v8h*)(p + 8);
  v16h r;
#pragma unroll
  for (int i = 0; i < 8; ++i) { r[i] = lo[i]; r[i + 8] = hi[i]; }
  return r;
}

// Per-wave 32(M) x 64(N) tile, K = 576.
// Each B fragment is reused by 2 A-row subtiles: 1.5 b128 loads per WMMA.
__device__ __forceinline__ void wmma_tile_32x64(const _Float16* __restrict__ A,
                                                const _Float16* __restrict__ Bm,
                                                int m0, int n0, v8f acc[2][4]) {
  const int lane = threadIdx.x & 31;
  const int half = lane >> 4;
  const int l16  = lane & 15;
  const _Float16* ap = A  + (size_t)(m0 + l16) * KC + half * 16;
  const _Float16* bp = Bm + (size_t)(n0 + l16) * KC + half * 16;
#pragma unroll 2
  for (int kt = 0; kt < KC / 32; ++kt) {
    v16h a0 = frag_a(ap + kt * 32);
    v16h a1 = frag_a(ap + (size_t)16 * KC + kt * 32);
#pragma unroll
    for (int s = 0; s < 4; ++s) {
      v16h b = frag_b(bp + (size_t)s * 16 * KC + kt * 32);
      acc[0][s] = __builtin_amdgcn_wmma_f32_16x16x32_f16(
          false, a0, false, b, (short)0, acc[0][s], false, false);
      acc[1][s] = __builtin_amdgcn_wmma_f32_16x16x32_f16(
          false, a1, false, b, (short)0, acc[1][s], false, false);
    }
  }
}

// ---------------------------------------------------------------------------
// f32 -> f16 flat conversion (weights: [O][C][3][3] flattens to [M][K=ci*9+tap])
// ---------------------------------------------------------------------------
__global__ void dcn_cvt_f16(const float* __restrict__ src, _Float16* __restrict__ dst, int n) {
  int i = blockIdx.x * blockDim.x + threadIdx.x;
  if (i < n) dst[i] = (_Float16)src[i];
}

// ---------------------------------------------------------------------------
// im2col of feat into f16 [N=65536][K=576] (K contiguous), zero-padded 3x3.
// n fastest across threads -> coalesced feat reads, heavy tap reuse in cache.
// ---------------------------------------------------------------------------
__global__ void dcn_im2col(const float* __restrict__ feat, _Float16* __restrict__ X) {
  int g  = blockIdx.x * blockDim.x + threadIdx.x;   // NFEAT * NPIX threads
  int n  = g & (NPIX - 1);
  int ci = g >> 16;
  int b = n >> 14, hw = n & 16383, h = hw >> 7, w = hw & 127;
  const float* img = feat + (((size_t)b * NFEAT + ci) << 14);
  _Float16* row = X + (size_t)n * KC + ci * KTAP;
#pragma unroll
  for (int t = 0; t < 9; ++t) {
    int y = h + t / 3 - 1, x = w + t % 3 - 1;
    float v = (y >= 0 && y < HH && x >= 0 && x < WW) ? img[(y << 7) + x] : 0.0f;
    row[t] = (_Float16)v;
  }
}

// ---------------------------------------------------------------------------
// Offset/mask conv as implicit GEMM: M=1728, N=65536, K=576.
// Epilogue fuses bias add + sigmoid (mask channels), writes f16 OM[co][n].
// Sigmoid predicate is wave-uniform per 16-row subtile (OFFCH % 16 == 0).
// ---------------------------------------------------------------------------
__global__ void dcn_conv_gemm(const _Float16* __restrict__ Aw, const _Float16* __restrict__ X,
                              const float* __restrict__ bias, _Float16* __restrict__ OM) {
  const int MT  = MCONV / 32;                       // 54 M-tiles
  const int wid = blockIdx.x * (blockDim.x >> 5) + (threadIdx.x >> 5);
  if (wid >= MT * (NPIX / 64)) return;
  const int mt = wid % MT, nb = wid / MT;
  const int m0 = mt * 32, n0 = nb * 64;
  const int lane = threadIdx.x & 31, half = lane >> 4, l16 = lane & 15;

  v8f acc[2][4] = {};
  wmma_tile_32x64(Aw, X, m0, n0, acc);

  // C/D layout: VGPR r -> M = m0 + a*16 + r + 8*half ; N = n0 + s*16 + (lane&15)
#pragma unroll
  for (int a = 0; a < 2; ++a) {
    const bool is_mask = (m0 + a * 16) >= OFFCH;    // uniform per subtile
#pragma unroll
    for (int s = 0; s < 4; ++s) {
      int n = n0 + s * 16 + l16;
#pragma unroll
      for (int r = 0; r < 8; ++r) {
        int co = m0 + a * 16 + r + half * 8;
        float v = acc[a][s][r] + bias[co];
        if (is_mask) v = 1.0f / (1.0f + __expf(-v));
        OM[(size_t)co * NPIX + n] = (_Float16)v;
      }
    }
  }
}

// ---------------------------------------------------------------------------
// Deformable bilinear sampling. One thread per (c, pixel), loops 9 taps.
// Reads f16 offsets/masks (coalesced over n), writes f16 S[n][c*9+kk].
// ---------------------------------------------------------------------------
__device__ __forceinline__ float dcn_at(const float* __restrict__ img, int y, int x) {
  return (y >= 0 && y < HH && x >= 0 && x < WW) ? img[(y << 7) + x] : 0.0f;
}

__global__ void dcn_sample(const float* __restrict__ inputs, const _Float16* __restrict__ OM,
                           _Float16* __restrict__ S) {
  int g = blockIdx.x * blockDim.x + threadIdx.x;    // CIN * NPIX threads
  int n = g & (NPIX - 1);
  int c = g >> 16;
  int b = n >> 14, hw = n & 16383, h = hw >> 7, w = hw & 127;
  const float* img = inputs + (((size_t)b * CIN + c) << 14);
  _Float16* srow = S + (size_t)n * KC + c * KTAP;
#pragma unroll
  for (int kk = 0; kk < 9; ++kk) {
    int coy = (c * KTAP + kk) * 2;                  // interleaved (y,x) offsets
    float oy = (float)OM[(size_t)coy * NPIX + n];
    float ox = (float)OM[(size_t)(coy + 1) * NPIX + n];
    float m  = (float)OM[(size_t)(OFFCH + c * KTAP + kk) * NPIX + n];
    float py = (float)(h + kk / 3 - 1) + oy;
    float px = (float)(w + kk % 3 - 1) + ox;
    float y0f = floorf(py), x0f = floorf(px);
    float wy = py - y0f, wx = px - x0f;
    int y0 = (int)y0f, x0 = (int)x0f;
    float v00 = dcn_at(img, y0,     x0);
    float v01 = dcn_at(img, y0,     x0 + 1);
    float v10 = dcn_at(img, y0 + 1, x0);
    float v11 = dcn_at(img, y0 + 1, x0 + 1);
    float sv = ((1.0f - wy) * (1.0f - wx) * v00 + (1.0f - wy) * wx * v01 +
                wy * (1.0f - wx) * v10 + wy * wx * v11) * m;
    srow[kk] = (_Float16)sv;
  }
}

// ---------------------------------------------------------------------------
// Output GEMM: out[b,o,h,w] = relu( sum_k dcw[o][k]*S[n][k] + b[o] )
// M=64, N=65536, K=576. Writes f32 NCHW directly to d_out.
// ---------------------------------------------------------------------------
__global__ void dcn_out_gemm(const _Float16* __restrict__ Dw, const _Float16* __restrict__ S,
                             const float* __restrict__ bias, float* __restrict__ out) {
  const int MT  = COUT / 32;                        // 2 M-tiles
  const int wid = blockIdx.x * (blockDim.x >> 5) + (threadIdx.x >> 5);
  if (wid >= MT * (NPIX / 64)) return;
  const int mt = wid % MT, nb = wid / MT;
  const int m0 = mt * 32, n0 = nb * 64;
  const int lane = threadIdx.x & 31, half = lane >> 4, l16 = lane & 15;

  v8f acc[2][4] = {};
  wmma_tile_32x64(Dw, S, m0, n0, acc);

#pragma unroll
  for (int a = 0; a < 2; ++a) {
#pragma unroll
    for (int s = 0; s < 4; ++s) {
      int n = n0 + s * 16 + l16;
      int b = n >> 14, hw = n & 16383;
#pragma unroll
      for (int r = 0; r < 8; ++r) {
        int o = m0 + a * 16 + r + half * 8;
        float v = acc[a][s][r] + bias[o];
        v = fmaxf(v, 0.0f);
        out[(((size_t)b * COUT + o) << 14) + hw] = v;
      }
    }
  }
}

// ---------------------------------------------------------------------------
// Launch
// ---------------------------------------------------------------------------
extern "C" void kernel_launch(void* const* d_in, const int* in_sizes, int n_in,
                              void* d_out, int out_size, void* d_ws, size_t ws_size,
                              hipStream_t stream) {
  const float* inputs = (const float*)d_in[0];
  const float* feat   = (const float*)d_in[1];
  const float* om_w   = (const float*)d_in[2];
  const float* om_b   = (const float*)d_in[3];
  const float* dc_w   = (const float*)d_in[4];
  const float* dc_b   = (const float*)d_in[5];
  float* out = (float*)d_out;

  // Workspace layout (all 16B aligned): Aw | Dw | X(im2col) | OM | S  ~380 MB
  char* ws = (char*)d_ws;
  size_t off = 0;
  _Float16* Aw = (_Float16*)(ws + off); off += (size_t)MCONV * KC * sizeof(_Float16);
  _Float16* Dw = (_Float16*)(ws + off); off += (size_t)COUT  * KC * sizeof(_Float16);
  _Float16* X  = (_Float16*)(ws + off); off += (size_t)NPIX  * KC * sizeof(_Float16);
  _Float16* OM = (_Float16*)(ws + off); off += (size_t)MCONV * NPIX * sizeof(_Float16);
  _Float16* S  = (_Float16*)(ws + off);

  dcn_cvt_f16<<<(MCONV * KC + 255) / 256, 256, 0, stream>>>(om_w, Aw, MCONV * KC);
  dcn_cvt_f16<<<(COUT * KC + 255) / 256, 256, 0, stream>>>(dc_w, Dw, COUT * KC);
  dcn_im2col<<<(NFEAT * NPIX) / 256, 256, 0, stream>>>(feat, X);
  // 54 M-tiles * 1024 N-blocks = 55296 wave-tiles, 8 waves per 256-thread block
  dcn_conv_gemm<<<(54 * (NPIX / 64)) / 8, 256, 0, stream>>>(Aw, X, om_b, OM);
  dcn_sample<<<(CIN * NPIX) / 256, 256, 0, stream>>>(inputs, OM, S);
  // 2 M-tiles * 1024 N-blocks = 2048 wave-tiles
  dcn_out_gemm<<<(2 * (NPIX / 64)) / 8, 256, 0, stream>>>(Dw, S, dc_b, out);
}
